// RWKV_TimeMix_60550448939127
// MI455X (gfx1250) — matrix operand order repaired and checked by reading, not compiled, not run on hardware
//
#include <hip/hip_runtime.h>
#include <hip/hip_bf16.h>

#define B_    256
#define S_    169
#define C_    512
#define H_    256
#define TT_   256
#define HS_   16
#define OUT_  64
#define KROWS_ 176   // t-rows stored for cumsum input (ceil16(S))

typedef __attribute__((ext_vector_type(16))) __bf16 v16bf;
typedef __attribute__((ext_vector_type(8)))  float  v8f;
typedef unsigned int u32x4 __attribute__((ext_vector_type(4)));
typedef int          i32x4 __attribute__((ext_vector_type(4)));
typedef int          i32x8 __attribute__((ext_vector_type(8)));
typedef __attribute__((address_space(3))) float    lds_float;
typedef __attribute__((address_space(3))) unsigned lds_uint;

union BF16Frag {
  unsigned int u[8];
  uint4 q[2];
  v16bf v;
};

__device__ __forceinline__ unsigned short bf16_bits(float f) {
  union { float f; unsigned int u; } x; x.f = f;
  unsigned int u = x.u;
  u += 0x7fffu + ((u >> 16) & 1u);     // round-to-nearest-even
  return (unsigned short)(u >> 16);
}
__device__ __forceinline__ unsigned int pack_bf16(float lo, float hi) {
  return (unsigned int)bf16_bits(lo) | ((unsigned int)bf16_bits(hi) << 16);
}

#if __has_builtin(__builtin_amdgcn_tensor_load_to_lds) && __has_builtin(__builtin_amdgcn_s_wait_tensorcnt)
#define HAVE_TDM 1
#else
#define HAVE_TDM 0
#endif

#if HAVE_TDM
// 1-D contiguous TDM copy: ndw dwords from gsrc -> LDS byte address lds_base.
// D# per CDNA5 ISA ch.8: group0 {count=1, lds_addr, global_addr, type=2},
// group1 {data_size=4B, tensor_dim0=tile_dim0=ndw, tensor_dim1=tile_dim1=1,
//         dim0_stride=ndw}, groups 2/3 zero (<=2D tensor).
__device__ __forceinline__ void tdm_load_1d(unsigned lds_base, const void* gsrc,
                                            unsigned ndw) {
  unsigned long long ga = (unsigned long long)gsrc;
  u32x4 g0;
  g0.x = 1u;                                    // count=1 (valid), user mode
  g0.y = lds_base;                              // lds_addr [63:32]
  g0.z = (unsigned)ga;                          // global_addr [95:64]
  g0.w = (unsigned)(ga >> 32) | 0x80000000u;    // global_addr hi | type=2 <<30
  i32x8 g1;
  g1[0] = 0x20000;                              // data_size=2 (4B), mask=0
  g1[1] = (int)((ndw & 0xFFFFu) << 16);         // tensor_dim0 lo -> bits[63:48]
  g1[2] = (int)((ndw >> 16) | (1u << 16));      // tensor_dim0 hi | tensor_dim1=1
  g1[3] = (int)(ndw << 16);                     // tile_dim0 -> bits[127:112]
  g1[4] = 1;                                    // tile_dim1=1, tile_dim2=0
  g1[5] = (int)ndw;                             // tensor_dim0_stride lo
  g1[6] = 0;                                    // stride hi, dim1_stride lo
  g1[7] = 0;
  i32x4 z4 = {0, 0, 0, 0};
#if __clang_major__ >= 23
  i32x8 z8 = {0, 0, 0, 0, 0, 0, 0, 0};
  __builtin_amdgcn_tensor_load_to_lds(g0, g1, z4, z4, z8, 0);
#else
  __builtin_amdgcn_tensor_load_to_lds(g0, g1, z4, z4, 0);
#endif
}
#endif

// ---------------------------------------------------------------------------
// Kernel W: pre-pack KVR projection weights [48 x 512] f32 -> bf16 WMMA
// B-fragments (16 K-chunks x 3 N-tiles) once, into workspace. Blocks in
// kernel B then TDM-copy the 48KB image straight into LDS.
// B 32x16 layout: lane = col (lanes>=16 hold K 16..31), K contiguous per lane.
// ---------------------------------------------------------------------------
__global__ void rwkv_wpack_kernel(const float* __restrict__ key_w,
                                  const float* __restrict__ value_w,
                                  const float* __restrict__ recep_w,
                                  unsigned int* __restrict__ wfrag) {
  const int e = blockIdx.x * 256 + threadIdx.x;   // 0..1535 = (frag, lane)
  const int fragIdx = e >> 5;                     // kc*3 + nt
  const int l = e & 31;
  const int kc = fragIdx / 3;
  const int nt = fragIdx % 3;
  const int col = nt * 16 + (l & 15);
  const float* wrow = (col < 16) ? (key_w + col * C_)
                    : (col < 32) ? (value_w + (col - 16) * C_)
                                 : (recep_w + (col - 32) * C_);
  const int c0 = kc * 32 + ((l >= 16) ? 16 : 0);
  float4 f0 = *(const float4*)(wrow + c0);
  float4 f1 = *(const float4*)(wrow + c0 + 4);
  float4 f2 = *(const float4*)(wrow + c0 + 8);
  float4 f3 = *(const float4*)(wrow + c0 + 12);
  unsigned int* dst = wfrag + (fragIdx * 32 + l) * 8;
  dst[0] = pack_bf16(f0.x, f0.y); dst[1] = pack_bf16(f0.z, f0.w);
  dst[2] = pack_bf16(f1.x, f1.y); dst[3] = pack_bf16(f1.z, f1.w);
  dst[4] = pack_bf16(f2.x, f2.y); dst[5] = pack_bf16(f2.z, f2.w);
  dst[6] = pack_bf16(f3.x, f3.y); dst[7] = pack_bf16(f3.z, f3.w);
}

// ---------------------------------------------------------------------------
// Kernel A: w2[h,t] = beta[h,t] * sum_{u<=t} time_w[h, TT-1-t+u] * alpha[h,u]
// written as bf16 directly into WMMA A-fragment layout (A 16x32 bf16).
// ---------------------------------------------------------------------------
__global__ void rwkv_w2_kernel(const float* __restrict__ time_w,
                               const float* __restrict__ alpha,
                               const float* __restrict__ beta,
                               unsigned short* __restrict__ w2t_frag) {
  const int h = blockIdx.x;     // K dimension of wkv GEMM
  const int t = threadIdx.x;    // M dimension
  const float* wrow = time_w + h * TT_;
  const float* arow = alpha  + h * TT_;   // time_alpha [H,1,TT]
  float s = 0.f;
  const int base = TT_ - 1 - t;
  for (int u = 0; u <= t; ++u) s += wrow[base + u] * arow[u];
  s *= beta[h * TT_ + t];                  // time_beta [H,TT,1]

  const int mt = t >> 4, row = t & 15;
  const int kc = h >> 5, ko = h & 31;
  const int lane = row + ((ko & 16) ? 16 : 0);
  const int pos  = ko & 15;
  w2t_frag[(((mt * 8 + kc) * 32) + lane) * 16 + pos] = bf16_bits(s);
}

// ---------------------------------------------------------------------------
// Kernel B: per batch b, fused time-shift + K/V/R projection GEMM
//   [176 rows x 512] x [512 x 48]  via v_wmma_f32_16x16x32_bf16
// Weight fragments arrive in LDS via one TDM tensor_load_to_lds (48 KB).
// ---------------------------------------------------------------------------
__launch_bounds__(256, 2)
__global__ void rwkv_kvr_kernel(const float* __restrict__ x,
                                const unsigned int* __restrict__ wfrag,
                                const float* __restrict__ key_b,
                                const float* __restrict__ value_b,
                                const float* __restrict__ recep_b,
                                float* __restrict__ k_ws, float* __restrict__ kv_ws,
                                float* __restrict__ r_ws) {
  __shared__ unsigned int bfr[48 * 32 * 8];     // 48 KB of weight B-fragments

  const int b    = blockIdx.x;
  const int tid  = threadIdx.x;
  const int lane = tid & 31;
  const int wave = tid >> 5;

#if HAVE_TDM
  if (wave == 0) {                      // TDM is wave-level; one wave drives DMA
    unsigned lds_base = (unsigned)(unsigned long long)(lds_uint*)&bfr[0];
    tdm_load_1d(lds_base, wfrag, 48 * 32 * 8);
    __builtin_amdgcn_s_wait_tensorcnt(0);
  }
#else
  #pragma unroll
  for (int q = 0; q < 12; ++q)
    ((uint4*)bfr)[tid + q * 256] = ((const uint4*)wfrag)[tid + q * 256];
#endif
  __syncthreads();

  const int  hi = (lane >= 16) ? 1 : 0;
  const int  c  = lane & 15;
  const float kb = key_b[c], vb = value_b[c], rb = recep_b[c];
  const uint4* ldsq = (const uint4*)bfr;

  #pragma unroll 1
  for (int mi = 0; mi < 2; ++mi) {
    const int m = wave + mi * 8;              // 16 M-tiles over 8 waves
    if (m * 16 < S_) {
      v8f accK = {}, accV = {}, accR = {};
      const int rr = lane & 15;               // A-matrix row within tile
      const int t  = m * 16 + rr;
      #pragma unroll 1
      for (int kc = 0; kc < 16; ++kc) {
        // time shift: channels <256 (kc<8) read row t-1, else row t
        const int srcT = (kc < 8) ? (t - 1) : t;
        const bool valid = (t < S_) && (srcT >= 0);
        const int c0 = kc * 32 + (hi ? 16 : 0);   // A lane K contiguous, 16 per lane half
        BF16Frag a;
        if (valid) {
          const float* xr = x + ((size_t)b * S_ + (size_t)srcT) * C_ + c0;
          float4 f0 = *(const float4*)(xr);
          float4 f1 = *(const float4*)(xr + 4);
          float4 f2 = *(const float4*)(xr + 8);
          float4 f3 = *(const float4*)(xr + 12);
          a.u[0] = pack_bf16(f0.x, f0.y); a.u[1] = pack_bf16(f0.z, f0.w);
          a.u[2] = pack_bf16(f1.x, f1.y); a.u[3] = pack_bf16(f1.z, f1.w);
          a.u[4] = pack_bf16(f2.x, f2.y); a.u[5] = pack_bf16(f2.z, f2.w);
          a.u[6] = pack_bf16(f3.x, f3.y); a.u[7] = pack_bf16(f3.z, f3.w);
          if (kc + 1 < 16) __builtin_prefetch(xr + 32, 0, 1);  // global_prefetch_b8
        } else {
          #pragma unroll
          for (int q = 0; q < 8; ++q) a.u[q] = 0u;
        }
        BF16Frag bk, bv, br;
        bk.q[0] = ldsq[((kc * 3 + 0) * 32 + lane) * 2 + 0];
        bk.q[1] = ldsq[((kc * 3 + 0) * 32 + lane) * 2 + 1];
        bv.q[0] = ldsq[((kc * 3 + 1) * 32 + lane) * 2 + 0];
        bv.q[1] = ldsq[((kc * 3 + 1) * 32 + lane) * 2 + 1];
        br.q[0] = ldsq[((kc * 3 + 2) * 32 + lane) * 2 + 0];
        br.q[1] = ldsq[((kc * 3 + 2) * 32 + lane) * 2 + 1];
        accK = __builtin_amdgcn_wmma_f32_16x16x32_bf16(false, a.v, false, bk.v, (short)0, accK, false, false);
        accV = __builtin_amdgcn_wmma_f32_16x16x32_bf16(false, a.v, false, bv.v, (short)0, accV, false, false);
        accR = __builtin_amdgcn_wmma_f32_16x16x32_bf16(false, a.v, false, br.v, (short)0, accR, false, false);
      }
      #pragma unroll
      for (int i = 0; i < 8; ++i) {
        const int tt = m * 16 + i + (hi ? 8 : 0);
        const bool tv = tt < S_;
        float kx = 0.f, kvv = 0.f, rv = 0.f;
        if (tv) {
          kx  = __expf(fminf(fmaxf(accK[i] + kb, -60.f), 30.f));
          kvv = kx * (accV[i] + vb);
          rv  = accR[i] + rb;
        }
        k_ws [((size_t)b * KROWS_ + tt) * HS_ + c] = kx;
        kv_ws[((size_t)b * TT_    + tt) * HS_ + c] = kvv;
        r_ws [((size_t)b * TT_    + tt) * HS_ + c] = rv;
      }
    } else {
      #pragma unroll
      for (int i = 0; i < 8; ++i) {
        const int tt = m * 16 + i + (hi ? 8 : 0);
        kv_ws[((size_t)b * TT_ + tt) * HS_ + c] = 0.f;
        r_ws [((size_t)b * TT_ + tt) * HS_ + c] = 0.f;
      }
    }
  }
}

// ---------------------------------------------------------------------------
// Kernel C: denom = cumsum_t(k) for t<S else 1.0   (4096 independent lanes)
// ---------------------------------------------------------------------------
__global__ void rwkv_cumsum_kernel(const float* __restrict__ k_ws,
                                   float* __restrict__ den_ws) {
  const int id = blockIdx.x * blockDim.x + threadIdx.x;   // 0..4095
  const int b = id >> 4, c = id & 15;
  float acc = 0.f;
  for (int t = 0; t < TT_; ++t) {
    if (t < S_) acc += k_ws[((size_t)b * KROWS_ + t) * HS_ + c];
    den_ws[((size_t)b * TT_ + t) * HS_ + c] = (t < S_) ? acc : 1.0f;
  }
}

// ---------------------------------------------------------------------------
// Kernel D: per batch b:
//   wkv = W2^T [256x256] @ kv[b] [256x16]   (bf16 WMMA, f32 acc)
//   rwkv = sigmoid(r) * wkv / denom
//   out  = (rwkv @ out_w^T + out_b) * gamma[t]   (second WMMA, K=16 padded)
// kv[b] (16 KB) staged into LDS with global_load_async_to_lds_b128.
// ---------------------------------------------------------------------------
__launch_bounds__(256, 2)
__global__ void rwkv_wkv_out_kernel(const unsigned short* __restrict__ w2t_frag,
                                    const float* __restrict__ kv_ws,
                                    const float* __restrict__ r_ws,
                                    const float* __restrict__ den_ws,
                                    const float* __restrict__ out_w,
                                    const float* __restrict__ out_b,
                                    const float* __restrict__ gamma,
                                    float* __restrict__ out) {
  __shared__ float kvs[TT_ * HS_];         // kv[b], contiguous for async b128
  __shared__ float tileLds[8][16 * 17];    // per-wave rwkv tile (padded)

  const int b    = blockIdx.x;
  const int tid  = threadIdx.x;
  const int lane = tid & 31;
  const int wave = tid >> 5;
  const int hi   = (lane >= 16) ? 1 : 0;
  const int c    = lane & 15;

  // async stage kv[b] (256 rows x 64B): each thread copies its row, 4x b128
  {
    const float* src = kv_ws + ((size_t)b * TT_ + tid) * HS_;
    unsigned lds_off = (unsigned)(unsigned long long)(lds_float*)&kvs[tid * HS_];
    unsigned long long ga = (unsigned long long)src;
    asm volatile("global_load_async_to_lds_b128 %0, %1, off\n\t"
                 "global_load_async_to_lds_b128 %0, %1, off offset:16\n\t"
                 "global_load_async_to_lds_b128 %0, %1, off offset:32\n\t"
                 "global_load_async_to_lds_b128 %0, %1, off offset:48"
                 :: "v"(lds_off), "v"(ga) : "memory");
    asm volatile("s_wait_asynccnt 0" ::: "memory");
  }
  __syncthreads();

  // out-projection B-fragments: B[k=c][n=o] = out_w[o][c]; K 16..31 zero
  BF16Frag ob[4];
  float obias[4];
  #pragma unroll
  for (int nt = 0; nt < 4; ++nt) {
    if (!hi) {
      const float* wr = out_w + (size_t)(nt * 16 + c) * HS_;
      float4 f0 = *(const float4*)(wr);
      float4 f1 = *(const float4*)(wr + 4);
      float4 f2 = *(const float4*)(wr + 8);
      float4 f3 = *(const float4*)(wr + 12);
      ob[nt].u[0] = pack_bf16(f0.x, f0.y); ob[nt].u[1] = pack_bf16(f0.z, f0.w);
      ob[nt].u[2] = pack_bf16(f1.x, f1.y); ob[nt].u[3] = pack_bf16(f1.z, f1.w);
      ob[nt].u[4] = pack_bf16(f2.x, f2.y); ob[nt].u[5] = pack_bf16(f2.z, f2.w);
      ob[nt].u[6] = pack_bf16(f3.x, f3.y); ob[nt].u[7] = pack_bf16(f3.z, f3.w);
    } else {
      #pragma unroll
      for (int q = 0; q < 8; ++q) ob[nt].u[q] = 0u;   // K 16..31 of pad
    }
    obias[nt] = out_b[nt * 16 + c];
  }

  #pragma unroll 1
  for (int mi = 0; mi < 2; ++mi) {
    const int m = wave + mi * 8;
    v8f acc = {};
    #pragma unroll 1
    for (int kc = 0; kc < 8; ++kc) {
      BF16Frag a;   // precomputed w2^T A-fragment: 2 coalesced b128 loads
      const uint4* ap = (const uint4*)(w2t_frag) + (size_t)((m * 8 + kc) * 32 + lane) * 2;
      a.q[0] = ap[0]; a.q[1] = ap[1];
      BF16Frag bf;  // kv B-fragment: lane = col c, K = h contiguous
      const int h0 = kc * 32 + (hi ? 16 : 0);
      #pragma unroll
      for (int p = 0; p < 8; ++p)
        bf.u[p] = pack_bf16(kvs[(h0 + 2 * p) * HS_ + c], kvs[(h0 + 2 * p + 1) * HS_ + c]);
      acc = __builtin_amdgcn_wmma_f32_16x16x32_bf16(false, a.v, false, bf.v, (short)0, acc, false, false);
    }

    // rwkv = sigmoid(r) * wkv / denom  -> per-wave LDS tile (in-order DS)
    float* tl = tileLds[wave];
    #pragma unroll
    for (int i = 0; i < 8; ++i) {
      const int t = m * 16 + i + (hi ? 8 : 0);
      const size_t idx = ((size_t)b * TT_ + t) * HS_ + c;
      const float rv = r_ws[idx];
      const float dn = den_ws[idx];
      const float sg = 1.f / (1.f + __expf(-rv));
      tl[(i + (hi ? 8 : 0)) * 17 + c] = sg * acc[i] / dn;
    }

    // A-fragment for out projection (K=16, upper half zero)
    BF16Frag af;
    {
      const int row = c;
      #pragma unroll
      for (int p = 0; p < 8; ++p)
        af.u[p] = hi ? 0u : pack_bf16(tl[row * 17 + 2 * p], tl[row * 17 + 2 * p + 1]);
    }

    #pragma unroll
    for (int nt = 0; nt < 4; ++nt) {
      v8f o = {};
      o = __builtin_amdgcn_wmma_f32_16x16x32_bf16(false, af.v, false, ob[nt].v, (short)0, o, false, false);
      #pragma unroll
      for (int i = 0; i < 8; ++i) {
        const int t = m * 16 + i + (hi ? 8 : 0);
        out[((size_t)b * TT_ + t) * OUT_ + nt * 16 + c] = (o[i] + obias[nt]) * gamma[t];
      }
    }
  }
}

// ---------------------------------------------------------------------------
extern "C" void kernel_launch(void* const* d_in, const int* in_sizes, int n_in,
                              void* d_out, int out_size, void* d_ws, size_t ws_size,
                              hipStream_t stream) {
  const float* x          = (const float*)d_in[0];
  const float* time_w     = (const float*)d_in[1];
  const float* time_alpha = (const float*)d_in[2];
  const float* time_beta  = (const float*)d_in[3];
  const float* time_gamma = (const float*)d_in[4];
  const float* key_w      = (const float*)d_in[5];
  const float* key_b      = (const float*)d_in[6];
  const float* value_w    = (const float*)d_in[7];
  const float* value_b    = (const float*)d_in[8];
  const float* recep_w    = (const float*)d_in[9];
  const float* recep_b    = (const float*)d_in[10];
  const float* out_w      = (const float*)d_in[11];
  const float* out_b      = (const float*)d_in[12];

  char* ws = (char*)d_ws;
  const size_t W2T_BYTES   = (size_t)TT_ * H_ * 2;            // 128 KB bf16 frags
  const size_t WFRAG_BYTES = (size_t)48 * 32 * 8 * 4;         // 48 KB weight frags
  const size_t KV_BYTES    = (size_t)B_ * TT_ * HS_ * 4;      // 4 MB
  const size_t R_BYTES     = (size_t)B_ * TT_ * HS_ * 4;      // 4 MB
  const size_t K_BYTES     = (size_t)B_ * KROWS_ * HS_ * 4;   // ~2.9 MB
  unsigned short* w2t   = (unsigned short*)(ws);
  unsigned int*   wfrag = (unsigned int*)(ws + W2T_BYTES);
  float* kv_ws  = (float*)(ws + W2T_BYTES + WFRAG_BYTES);
  float* r_ws   = (float*)(ws + W2T_BYTES + WFRAG_BYTES + KV_BYTES);
  float* k_ws   = (float*)(ws + W2T_BYTES + WFRAG_BYTES + KV_BYTES + R_BYTES);
  float* den_ws = (float*)(ws + W2T_BYTES + WFRAG_BYTES + KV_BYTES + R_BYTES + K_BYTES);

  rwkv_wpack_kernel<<<6, 256, 0, stream>>>(key_w, value_w, recep_w, wfrag);
  rwkv_w2_kernel<<<H_, TT_, 0, stream>>>(time_w, time_alpha, time_beta, w2t);
  rwkv_kvr_kernel<<<B_, 256, 0, stream>>>(x, wfrag, key_b, value_b, recep_b,
                                          k_ws, kv_ws, r_ws);
  rwkv_cumsum_kernel<<<16, 256, 0, stream>>>(k_ws, den_ws);
  rwkv_wkv_out_kernel<<<B_, 256, 0, stream>>>(w2t, kv_ws, r_ws, den_ws,
                                              out_w, out_b, time_gamma, (float*)d_out);
}
